// PaiNN_25082609009456
// MI455X (gfx1250) — matrix operand order repaired and checked by reading, not compiled, not run on hardware
//
#include <hip/hip_runtime.h>
#include <hip/hip_bf16.h>

// ---------------------------------------------------------------------------
// PaiNN fused block for MI455X (gfx1250, wave32, WMMA).
//
// The reference's jnp.sum() reductions are GLOBAL scalars, so
//   s_m = c_s + s_j,  v_m = c_v + v_j,  v_m@W = v_j@W + c_v*colsum(W).
// => two streaming passes over the edge arrays (~0.9 GB HBM => ~40us roof),
//    all GEMMs via v_wmma_f32_16x16x32_f16 (f32 accum), weights staged to LDS
//    once per block with GLOBAL_LOAD_ASYNC_TO_LDS_B128 (ASYNCcnt path), and
//    A-fragments hoisted into registers across all N-tiles.
// ---------------------------------------------------------------------------

typedef __attribute__((ext_vector_type(16))) _Float16 v16h;
typedef __attribute__((ext_vector_type(8)))  _Float16 v8h;
typedef __attribute__((ext_vector_type(4)))  _Float16 v4h;
typedef __attribute__((ext_vector_type(8)))  float    v8f;

#define E_TOT 262144
#define TILES (E_TOT / 16)
#define WAVES 8
#define TPB   (WAVES * 32)
#define NBLK  (TILES / WAVES)   // 2048 blocks, 8 edge-tiles (waves) per block

// Async global->LDS copy of one 16B chunk (gfx1250 GLOBAL_LOAD_ASYNC_TO_LDS,
// GVS addressing: SGPR64 base + VGPR32 offset; VDST VGPR holds LDS byte addr).
__device__ __forceinline__ void async_cp16(void* lds_dst, const void* gbase,
                                           unsigned goff) {
  unsigned lds_addr = (unsigned)(size_t)lds_dst;  // low 32 bits = LDS offset
  asm volatile("global_load_async_to_lds_b128 %0, %1, %2"
               :: "v"(lds_addr), "v"(goff), "s"(gbase)
               : "memory");
}
__device__ __forceinline__ void async_wait0() {
  asm volatile("s_wait_asynccnt 0x0" ::: "memory");
}

// Load the four K-step A fragments (16x128 activation tile) once.
__device__ __forceinline__ void load_afrag(const _Float16* A, int lane,
                                           v16h a[4]) {
  const int m  = lane & 15;
  const int hi = lane >> 4;
  const _Float16* arow = A + m * 128 + hi * 8; // K = k0+off+{0..7,16..23}
#pragma unroll
  for (int k = 0; k < 4; ++k) {
    v8h lo  = *(const v8h*)(arow + k * 32);
    v8h hi8 = *(const v8h*)(arow + k * 32 + 16);
    a[k] = __builtin_shufflevector(lo, hi8,
             0,1,2,3,4,5,6,7,8,9,10,11,12,13,14,15);
  }
}

// One 16x16 f32 output tile: 4 x (16x16x32) WMMA, A fragments in registers.
// Bt: LDS [N][128] f16 (transposed weight: Bt[n*128+k] = W[k][n]).
__device__ __forceinline__ v8f mmtile(const v16h a[4], const _Float16* Bt,
                                      int nbase, int lane) {
  v8f c = {};
  const _Float16* bcol = Bt + (nbase + (lane & 15)) * 128 + (lane >> 4) * 8;
#pragma unroll
  for (int k = 0; k < 4; ++k) {
    v8h blo = *(const v8h*)(bcol + k * 32);  // K = k0+off+{0..7,16..23}
    v8h bhi = *(const v8h*)(bcol + k * 32 + 16);
    v16h b = __builtin_shufflevector(blo, bhi,
             0,1,2,3,4,5,6,7,8,9,10,11,12,13,14,15);
    c = __builtin_amdgcn_wmma_f32_16x16x32_f16(false, a[k], false, b,
                                               (short)0, c, false, false);
  }
  return c;
}

// Stage a 16x128 f32 global tile into LDS as f16 (one wave).
__device__ __forceinline__ void stage16x128(const float* src, _Float16* dst,
                                            int lane) {
  const int row = lane >> 1;
  const int c0  = (lane & 1) * 64;
  const float4* s = (const float4*)(src + row * 128 + c0);
  v4h* d = (v4h*)(dst + row * 128 + c0);
#pragma unroll
  for (int i = 0; i < 16; ++i) {
    float4 t = s[i];
    v4h h;
    h.x = (_Float16)t.x; h.y = (_Float16)t.y;
    h.z = (_Float16)t.z; h.w = (_Float16)t.w;
    d[i] = h;
  }
}

// ---------- prep kernels -----------------------------------------------------

__global__ void k_transpose(const float* __restrict__ src,  // [128 x N]
                            _Float16* __restrict__ dst,     // [N x 128]
                            int N) {
  int idx = blockIdx.x * blockDim.x + threadIdx.x;
  if (idx < 128 * N) {
    int k = idx / N, n = idx % N;
    dst[n * 128 + k] = (_Float16)src[idx];
  }
}

__global__ void k_colsum(const float* __restrict__ src,  // [128 x N]
                         float* __restrict__ dst, int N) {
  int n = blockIdx.x * blockDim.x + threadIdx.x;
  if (n < N) {
    float s = 0.f;
    for (int k = 0; k < 128; ++k) s += src[k * N + n];
    dst[n] = s;
  }
}

// ---------- phase 1: message block + scalar partial sums ---------------------

__global__ void __launch_bounds__(TPB)
painn_msg(const float* __restrict__ input1, const float* __restrict__ input2,
          const float* __restrict__ r_ij,   const float* __restrict__ v_j,
          const float* __restrict__ v_norm,
          const _Float16* __restrict__ W1t, const _Float16* __restrict__ W2t,
          const float* __restrict__ b1, const float* __restrict__ b2,
          const float* __restrict__ Wr, const float* __restrict__ br,
          float* __restrict__ partials) {
  extern __shared__ __attribute__((aligned(256))) char smem[];
  _Float16* sW1 = (_Float16*)smem;                 // 128*128
  _Float16* sW2 = sW1 + 128 * 128;                 // 384*128
  _Float16* sX  = sW2 + 384 * 128;                 // WAVES * 16*128
  _Float16* sA2 = sX  + WAVES * 2048;              // WAVES * 16*128
  float* sI2 = (float*)(sA2 + WAVES * 2048);       // WAVES * 16
  float* sFC = sI2 + WAVES * 16;                   // WAVES * 16
  float* red = sFC + WAVES * 16;                   // 2 * TPB

  const int tid = threadIdx.x;
  // Async-stage W1t||W2t (contiguous in workspace) into LDS, 16B per lane/iter.
  for (int i = tid; i < (128 * 128 + 384 * 128) * 2 / 16; i += TPB)
    async_cp16((char*)sW1 + i * 16, W1t, (unsigned)i * 16u);

  const int wave = tid >> 5, lane = tid & 31;
  const int e0 = (blockIdx.x * WAVES + wave) * 16;
  _Float16* myX  = sX  + wave * 2048;
  _Float16* myA2 = sA2 + wave * 2048;
  float* myI2 = sI2 + wave * 16;
  float* myFC = sFC + wave * 16;

  stage16x128(input1 + (size_t)e0 * 128, myX, lane);
  if (lane < 16) {
    int e = e0 + lane;
    myI2[lane] = input2[e];
    // f_c = 0.5*cos(pi*r/5) + 1
    myFC[lane] = 0.5f * __cosf(0.62831853071795864769f * r_ij[e]) + 1.0f;
  }
  async_wait0();
  __syncthreads();

  const int n  = lane & 15;
  const int mo = (lane >> 4) * 8;

  v16h af[4];
  load_afrag(myX, lane, af);

  // H1 = silu(X @ W1 + b1) -> A2 (f16, A-tile layout in LDS)
#pragma unroll
  for (int nt = 0; nt < 8; ++nt) {
    v8f c = mmtile(af, sW1, nt * 16, lane);
    int h = nt * 16 + n;
    float bias = b1[h];
#pragma unroll
    for (int r = 0; r < 8; ++r) {
      float x = c[r] + bias;
      myA2[(r + mo) * 128 + h] = (_Float16)(x / (1.0f + __expf(-x)));
    }
  }
  __syncthreads();

  load_afrag(myA2, lane, af);

  // M = (A2 @ W2 + b2) * w_out ; accumulate c_s (cols 128..255) and
  // c_v = sum(m0*v_j + m2*v_norm).
  float cs = 0.f, cv = 0.f;
  for (int nt = 0; nt < 8; ++nt) {           // m0 columns 0..127
    v8f c = mmtile(af, sW2, nt * 16, lane);
    int h = nt * 16 + n;
    float b2v = b2[h], wr = Wr[h], brv = br[h];
#pragma unroll
    for (int r = 0; r < 8; ++r) {
      int M = r + mo;
      float mval = (c[r] + b2v) * ((myI2[M] * wr + brv) * myFC[M]);
      cv += mval * v_j[(size_t)(e0 + M) * 128 + h];
    }
  }
  for (int nt = 8; nt < 16; ++nt) {          // m1 columns 128..255
    v8f c = mmtile(af, sW2, nt * 16, lane);
    int h = nt * 16 + n;
    float b2v = b2[h], wr = Wr[h], brv = br[h];
#pragma unroll
    for (int r = 0; r < 8; ++r) {
      int M = r + mo;
      cs += (c[r] + b2v) * ((myI2[M] * wr + brv) * myFC[M]);
    }
  }
  for (int nt = 16; nt < 24; ++nt) {         // m2 columns 256..383
    v8f c = mmtile(af, sW2, nt * 16, lane);
    int h = nt * 16 + n;
    float b2v = b2[h], wr = Wr[h], brv = br[h];
#pragma unroll
    for (int r = 0; r < 8; ++r) {
      int M = r + mo;
      float mval = (c[r] + b2v) * ((myI2[M] * wr + brv) * myFC[M]);
      cv += mval * v_norm[(size_t)(e0 + M) * 128 + (h - 256)];
    }
  }

  // Deterministic block tree reduction -> per-block partials.
  red[tid] = cs;
  red[TPB + tid] = cv;
  __syncthreads();
  for (int s = TPB / 2; s > 0; s >>= 1) {
    if (tid < s) {
      red[tid] += red[tid + s];
      red[TPB + tid] += red[TPB + tid + s];
    }
    __syncthreads();
  }
  if (tid == 0) {
    partials[blockIdx.x * 2 + 0] = red[0];
    partials[blockIdx.x * 2 + 1] = red[TPB];
  }
}

// ---------- phase 1.5: deterministic scalar reduction ------------------------

__global__ void painn_reduce(const float* __restrict__ partials, int nblocks,
                             float* __restrict__ scal) {
  __shared__ float rs[256], rv[256];
  int t = threadIdx.x;
  float a = 0.f, b = 0.f;
  for (int i = t; i < nblocks; i += 256) {
    a += partials[2 * i];
    b += partials[2 * i + 1];
  }
  rs[t] = a; rv[t] = b;
  __syncthreads();
  for (int s = 128; s > 0; s >>= 1) {
    if (t < s) { rs[t] += rs[t + s]; rv[t] += rv[t + s]; }
    __syncthreads();
  }
  if (t == 0) { scal[0] = rs[0]; scal[1] = rv[0]; }
}

// ---------- phase 2: update block, fused combine, final outputs --------------

__global__ void __launch_bounds__(TPB)
painn_upd(const float* __restrict__ v_j, const float* __restrict__ s_j,
          const _Float16* __restrict__ Wut, const _Float16* __restrict__ Wvt,
          const _Float16* __restrict__ Wst,
          const float* __restrict__ bu, const float* __restrict__ bv,
          const float* __restrict__ bs,
          const float* __restrict__ csWu, const float* __restrict__ csWv,
          const float* __restrict__ csWs,
          const float* __restrict__ scal,
          float* __restrict__ v_u, float* __restrict__ s_u) {
  extern __shared__ __attribute__((aligned(256))) char smem[];
  _Float16* sWu = (_Float16*)smem;          // 128*128
  _Float16* sWv = sWu + 128 * 128;          // 64*128
  _Float16* sWs = sWv + 64 * 128;           // 384*128
  _Float16* sAv = sWs + 384 * 128;          // WAVES * 16*128
  _Float16* sAs = sAv + WAVES * 2048;       // WAVES * 16*128
  float* sV = (float*)(sAs + WAVES * 2048); // WAVES * 16*64

  const int tid = threadIdx.x;
  // Async-stage Wut||Wvt||Wst (contiguous in workspace) into LDS.
  for (int i = tid; i < (128 * 128 + 64 * 128 + 384 * 128) * 2 / 16; i += TPB)
    async_cp16((char*)sWu + i * 16, Wut, (unsigned)i * 16u);

  const float c_s = scal[0];
  const float c_v = scal[1];

  const int wave = tid >> 5, lane = tid & 31;
  const int e0 = (blockIdx.x * WAVES + wave) * 16;
  _Float16* myAv = sAv + wave * 2048;
  _Float16* myAs = sAs + wave * 2048;
  float* myV = sV + wave * 16 * 64;

  stage16x128(v_j + (size_t)e0 * 128, myAv, lane);
  stage16x128(s_j + (size_t)e0 * 128, myAs, lane);
  async_wait0();
  __syncthreads();

  const int n  = lane & 15;
  const int mo = (lane >> 4) * 8;

  v16h av[4], as_[4];
  load_afrag(myAv, lane, av);
  load_afrag(myAs, lane, as_);

  // V = v_m @ Wv + bv = v_j@Wv + c_v*colsum(Wv) + bv   (16x64, kept in LDS)
#pragma unroll
  for (int j = 0; j < 4; ++j) {
    v8f c = mmtile(av, sWv, j * 16, lane);
    int h = j * 16 + n;
    float add = bv[h] + c_v * csWv[h];
#pragma unroll
    for (int r = 0; r < 8; ++r) myV[(r + mo) * 64 + h] = c[r] + add;
  }
  __syncthreads();

  for (int j = 0; j < 8; ++j) {
    const int h = j * 16 + n;                 // output column 0..127
    // U tile and S0 tile
    v8f cu = mmtile(av,  sWu, j * 16, lane);
    v8f c0 = mmtile(as_, sWs, j * 16, lane);
    const float ub  = bu[h] + c_v * csWu[h];
    const float s0b = bs[h] + c_s * csWs[h];
#pragma unroll
    for (int r = 0; r < 8; ++r) {
      int M = r + mo;
      size_t idx = (size_t)(e0 + M) * 128 + h;
      float U  = cu[r] + ub;
      float S0 = c0[r] + s0b;
      v_u[idx] = S0 * U + c_v + v_j[idx];     // out4 + v_m
    }
    // S1, S2 tiles share the output column hh = h
    v8f c1 = mmtile(as_, sWs, 128 + j * 16, lane);
    v8f c2 = mmtile(as_, sWs, 256 + j * 16, lane);
    const float s1b = bs[128 + h] + c_s * csWs[128 + h];
    const float s2b = bs[256 + h] + c_s * csWs[256 + h];
    const int hv = (j & 3) * 16 + n;          // V_dup index = h % 64
#pragma unroll
    for (int r = 0; r < 8; ++r) {
      int M = r + mo;
      size_t idx = (size_t)(e0 + M) * 128 + h;
      float S1 = c1[r] + s1b;
      float S2 = c2[r] + s2b;
      s_u[idx] = S2 + S1 * myV[M * 64 + hv] + c_s + s_j[idx];  // out6 + s_m
    }
  }
}

// ---------------------------------------------------------------------------

extern "C" void kernel_launch(void* const* d_in, const int* in_sizes, int n_in,
                              void* d_out, int out_size, void* d_ws,
                              size_t ws_size, hipStream_t stream) {
  const float* input1 = (const float*)d_in[0];
  const float* input2 = (const float*)d_in[1];
  const float* r_ij   = (const float*)d_in[2];
  const float* v_j    = (const float*)d_in[3];
  const float* s_j    = (const float*)d_in[4];
  const float* v_norm = (const float*)d_in[5];
  const float* W1 = (const float*)d_in[6];  const float* b1 = (const float*)d_in[7];
  const float* W2 = (const float*)d_in[8];  const float* b2 = (const float*)d_in[9];
  const float* Wr = (const float*)d_in[10]; const float* br = (const float*)d_in[11];
  const float* Wu = (const float*)d_in[12]; const float* bu = (const float*)d_in[13];
  const float* Wv = (const float*)d_in[14]; const float* bv = (const float*)d_in[15];
  const float* Ws = (const float*)d_in[16]; const float* bs = (const float*)d_in[17];

  // Workspace layout (bytes). W1t..W2t contiguous; Wut..Wst contiguous.
  char* ws = (char*)d_ws;
  _Float16* W1t = (_Float16*)(ws + 0);        // 32768 B
  _Float16* W2t = (_Float16*)(ws + 32768);    // 98304 B
  _Float16* Wut = (_Float16*)(ws + 131072);   // 32768 B
  _Float16* Wvt = (_Float16*)(ws + 163840);   // 16384 B
  _Float16* Wst = (_Float16*)(ws + 180224);   // 98304 B
  float* csWu     = (float*)(ws + 278528);    // 512 B
  float* csWv     = (float*)(ws + 279040);    // 256 B
  float* csWs     = (float*)(ws + 279296);    // 1536 B
  float* partials = (float*)(ws + 280832);    // 2*NBLK*4 = 16384 B
  float* scal     = (float*)(ws + 297216);    // 8 B

  // Prep: f16-transposed weights + column sums.
  k_transpose<<<64,  256, 0, stream>>>(W1, W1t, 128);
  k_transpose<<<192, 256, 0, stream>>>(W2, W2t, 384);
  k_transpose<<<64,  256, 0, stream>>>(Wu, Wut, 128);
  k_transpose<<<32,  256, 0, stream>>>(Wv, Wvt, 64);
  k_transpose<<<192, 256, 0, stream>>>(Ws, Wst, 384);
  k_colsum<<<1, 128, 0, stream>>>(Wu, csWu, 128);
  k_colsum<<<1, 64,  0, stream>>>(Wv, csWv, 64);
  k_colsum<<<2, 256, 0, stream>>>(Ws, csWs, 384);

  // Phase 1: message block + partial scalar sums.
  size_t smem_msg = (size_t)(128 * 128 + 384 * 128 + 2 * WAVES * 2048) * 2 +
                    (size_t)(2 * WAVES * 16 + 2 * TPB) * 4;   // 199680 B
  painn_msg<<<NBLK, TPB, smem_msg, stream>>>(input1, input2, r_ij, v_j, v_norm,
                                             W1t, W2t, b1, b2, Wr, br, partials);

  painn_reduce<<<1, 256, 0, stream>>>(partials, NBLK, scal);

  // Phase 2: update block -> outputs (v_u then s_u, concatenated flat).
  float* v_u = (float*)d_out;
  float* s_u = v_u + (size_t)E_TOT * 128;
  size_t smem_upd = (size_t)(128 * 128 + 64 * 128 + 384 * 128 +
                             2 * WAVES * 2048) * 2 +
                    (size_t)(WAVES * 16 * 64) * 4;            // 245760 B
  painn_upd<<<NBLK, TPB, smem_upd, stream>>>(v_j, s_j, Wut, Wvt, Wst,
                                             bu, bv, bs, csWu, csWv, csWs,
                                             scal, v_u, s_u);
}
